// RDViTEncoder_3874060501745
// MI455X (gfx1250) — matrix operand order repaired and checked by reading, not compiled, not run on hardware
//
#include <hip/hip_runtime.h>
#include <hip/hip_bf16.h>
#include <stdint.h>

// ---------------------------------------------------------------------------
// Types for CDNA5 WMMA (wave32): v_wmma_f32_16x16x32_bf16
// ---------------------------------------------------------------------------
typedef __attribute__((ext_vector_type(16))) __bf16 v16bf;
typedef __attribute__((ext_vector_type(8)))  float  v8f;
typedef __attribute__((ext_vector_type(4)))  uint32_t u32x4;
typedef __attribute__((ext_vector_type(8)))  uint32_t u32x8;

union FragBF { v16bf v; uint4 q[2]; };

__device__ __forceinline__ unsigned short f2bf(float f) {
  union { float f; uint32_t u; } x; x.f = f;
  uint32_t r = x.u + 0x7FFFu + ((x.u >> 16) & 1u);   // round-to-nearest-even
  return (unsigned short)(r >> 16);
}

// ---------------------------------------------------------------------------
// Tensor Data Mover: DMA a 128-row x 64-element (bf16) 2D tile into LDS with
// 16B row padding (rows land at 72-short pitch). D# per
// CDNA5 ISA, 2-group form (VADDR2/3 disabled).
// Issued from a single lane; TDM ignores EXEC, tracked by TENSORcnt.
// ---------------------------------------------------------------------------
__device__ __forceinline__ void tdm_load_tile_2d(const void* gptr, uint32_t lds_off,
                                                 uint32_t strideElems) {
  uint64_t ga = (uint64_t)(uintptr_t)gptr;
  u32x4 g0;
  g0.x = 1u;                                          // count=1, user descriptor
  g0.y = lds_off;                                     // lds_addr (bytes)
  g0.z = (uint32_t)ga;                                // global_addr[31:0]
  g0.w = ((uint32_t)(ga >> 32) & 0x01FFFFFFu) | (2u << 30);  // addr[56:32] | type=2
  u32x8 g1;
  g1[0] = (1u << 16)        // data_size = 2 bytes
        | (1u << 20)        // pad_enable
        | (4u << 22)        // pad_interval: 32 DWORDs (= one 64-bf16 row)
        | (3u << 25);       // pad_amount: 4 DWORDs (= 8 bf16 -> 72-short pitch)
  g1[1] = (strideElems & 0xFFFFu) << 16;              // tensor_dim0[15:0]
  g1[2] = (strideElems >> 16) | (128u << 16);         // tensor_dim0[31:16] | dim1 lo
  g1[3] = (64u << 16);                                // dim1 hi | tile_dim0 = 64
  g1[4] = 128u;                                       // tile_dim1 = 128, tile_dim2 = 0
  g1[5] = strideElems;                                // tensor_dim0_stride[31:0]
  g1[6] = 0u;
  g1[7] = 0u;
  asm volatile("tensor_load_to_lds %0, %1" :: "s"(g0), "s"(g1) : "memory");
}

// ---------------------------------------------------------------------------
// Elementwise converts / im2col
// ---------------------------------------------------------------------------
__global__ void k_f32_to_bf16(const float* __restrict__ in,
                              unsigned short* __restrict__ out, int n) {
  int i = blockIdx.x * 256 + threadIdx.x;
  if (i < n) out[i] = f2bf(in[i]);
}

// Transpose-convert one K x N fp32 weight to N x K bf16 (B^T for the GEMM).
__global__ void k_convT_bf16(const float* __restrict__ in,
                             unsigned short* __restrict__ out, int K, int N) {
  __shared__ float tile[32][33];
  int kb = blockIdx.y * 32, nb = blockIdx.x * 32;
  int tx = threadIdx.x & 31, ty = threadIdx.x >> 5;   // ty = 0..7
#pragma unroll
  for (int r = 0; r < 32; r += 8)
    tile[ty + r][tx] = in[(size_t)(kb + ty + r) * N + nb + tx];
  __syncthreads();
#pragma unroll
  for (int r = 0; r < 32; r += 8)
    out[(size_t)(nb + ty + r) * K + kb + tx] = f2bf(tile[tx][ty + r]);
}

// img (B,3,512,512) -> A (8192 x 768) bf16, row = b*1024 + gy*32 + gx
__global__ void k_im2col(const float* __restrict__ img,
                         unsigned short* __restrict__ out) {
  int i = blockIdx.x * 256 + threadIdx.x;
  if (i >= 8192 * 768) return;
  int row = i / 768, col = i % 768;
  int b = row >> 10, g = row & 1023, gy = g >> 5, gx = g & 31;
  int ci = col >> 8, r = col & 255, py = r >> 4, px = r & 15;
  size_t src = (((size_t)b * 3 + ci) * 512 + gy * 16 + py) * 512 + gx * 16 + px;
  out[i] = f2bf(img[src]);
}

// ---------------------------------------------------------------------------
// LayerNorm over C=768, writes bf16 (GEMM A operand)
// ---------------------------------------------------------------------------
__global__ void k_layernorm_bf16(const float* __restrict__ x,
                                 const float* __restrict__ w,
                                 const float* __restrict__ b,
                                 unsigned short* __restrict__ out) {
  __shared__ float red[256];
  int row = blockIdx.x, t = threadIdx.x;
  const float* xr = x + (size_t)row * 768;
  float v0 = xr[t], v1 = xr[t + 256], v2 = xr[t + 512];
  red[t] = v0 + v1 + v2;
  __syncthreads();
  for (int o = 128; o > 0; o >>= 1) { if (t < o) red[t] += red[t + o]; __syncthreads(); }
  float mu = red[0] * (1.f / 768.f);
  __syncthreads();
  float d0 = v0 - mu, d1 = v1 - mu, d2 = v2 - mu;
  red[t] = d0 * d0 + d1 * d1 + d2 * d2;
  __syncthreads();
  for (int o = 128; o > 0; o >>= 1) { if (t < o) red[t] += red[t + o]; __syncthreads(); }
  float rs = rsqrtf(red[0] * (1.f / 768.f) + 1e-5f);
  unsigned short* orow = out + (size_t)row * 768;
  orow[t]       = f2bf(d0 * rs * w[t]       + b[t]);
  orow[t + 256] = f2bf(d1 * rs * w[t + 256] + b[t + 256]);
  orow[t + 512] = f2bf(d2 * rs * w[t + 512] + b[t + 512]);
}

// ---------------------------------------------------------------------------
// WMMA GEMM:  out = A(MxK bf16, row-major) @ Bt(NxK bf16, row-major = B^T)
// Block: 256 thr (8 waves), tile 128x128, K-tile 64, wave tile 64x32.
// LDS tiles staged by the Tensor Data Mover, double-buffered so the DMA of
// tile k+1 overlaps the WMMA pipeline on tile k (TDM completes in order per
// wave, so s_wait_tensorcnt<=2 guarantees the tile being consumed is ready).
// ---------------------------------------------------------------------------
template <int MODE>
__global__ void k_gemm_bf16(const unsigned short* __restrict__ A,
                            const unsigned short* __restrict__ Bt,
                            const float* __restrict__ bias,
                            const float* __restrict__ extra,
                            float* __restrict__ outF,
                            unsigned short* __restrict__ outH,
                            int M, int N, int K) {
  // [buf][A|B][128][72]
  __shared__ __attribute__((aligned(16))) unsigned short smem[4 * 128 * 72];
  const uint32_t tileB = 128u * 72u * 2u;             // bytes per tile
  const int t = threadIdx.x;
  const int mBase = blockIdx.y * 128, nBase = blockIdx.x * 128;
  const int wave = t >> 5, lane = t & 31;
  const int mW = (wave & 1) * 64, nW = (wave >> 1) * 32;
  const int lh = lane >> 4, ll = lane & 15;

  v8f zerov = {};
  v8f acc[4][2];
#pragma unroll
  for (int i = 0; i < 4; i++)
#pragma unroll
    for (int j = 0; j < 2; j++) acc[i][j] = zerov;

  // prologue: stage first K-tile into buffer 0
  if (t == 0) {
    tdm_load_tile_2d(A  + (size_t)mBase * K, 0u,    (uint32_t)K);
    tdm_load_tile_2d(Bt + (size_t)nBase * K, tileB, (uint32_t)K);
  }
  int cur = 0;
  for (int k0 = 0; k0 < K; k0 += 64) {
    if (t == 0) {
      if (k0 + 64 < K) {
        uint32_t nb = (uint32_t)(cur ^ 1) * 2u * tileB;
        tdm_load_tile_2d(A  + (size_t)mBase * K + k0 + 64, nb,         (uint32_t)K);
        tdm_load_tile_2d(Bt + (size_t)nBase * K + k0 + 64, nb + tileB, (uint32_t)K);
        __builtin_amdgcn_s_wait_tensorcnt(2);   // current pair done, next in flight
      } else {
        __builtin_amdgcn_s_wait_tensorcnt(0);   // last tile: drain
      }
    }
    __syncthreads();
    unsigned short (*Asm)[72] = (unsigned short (*)[72])(smem + cur * 2 * 128 * 72);
    unsigned short (*Bsm)[72] = (unsigned short (*)[72])(smem + cur * 2 * 128 * 72 + 128 * 72);
#pragma unroll
    for (int ks = 0; ks < 64; ks += 32) {
      FragBF bf[2];
#pragma unroll
      for (int j = 0; j < 2; j++) {
        int n = nW + j * 16 + ll;       // B 32x16: V0-3: K=lh*8.., V4-7: K=16+lh*8..
        bf[j].q[0] = *(const uint4*)&Bsm[n][ks + lh * 8];
        bf[j].q[1] = *(const uint4*)&Bsm[n][ks + 16 + lh * 8];
      }
#pragma unroll
      for (int i = 0; i < 4; i++) {
        int m = mW + i * 16 + ll;       // A 16x32: V0-3: K=lh*8.., V4-7: K=16+lh*8..
        FragBF af;
        af.q[0] = *(const uint4*)&Asm[m][ks + lh * 8];
        af.q[1] = *(const uint4*)&Asm[m][ks + 16 + lh * 8];
#pragma unroll
        for (int j = 0; j < 2; j++)
          acc[i][j] = __builtin_amdgcn_wmma_f32_16x16x32_bf16(
              false, af.v, false, bf[j].v, (short)0, acc[i][j], false, false);
      }
    }
    __syncthreads();   // all waves done reading buf[cur] before it is re-filled
    cur ^= 1;
  }
#pragma unroll
  for (int i = 0; i < 4; i++) {
#pragma unroll
    for (int j = 0; j < 2; j++) {
      int col = nBase + nW + j * 16 + ll;
      float bv = bias ? bias[col] : 0.f;
#pragma unroll
      for (int r = 0; r < 8; r++) {
        int row = mBase + mW + i * 16 + r + lh * 8;
        float v = acc[i][j][r] + bv;
        if (MODE == 1) {
          float c = 0.7978845608028654f * (v + 0.044715f * v * v * v);
          v = 0.5f * v * (1.f + tanhf(c));
        } else if (MODE == 2) {
          v += extra[(size_t)row * N + col];
        } else if (MODE == 3) {
          v += extra[(size_t)(row & 1023) * N + col];
        }
        if (outF) outF[(size_t)row * N + col] = v;
        if (outH) outH[(size_t)row * N + col] = f2bf(v);
      }
    }
  }
}

// ---------------------------------------------------------------------------
// Window attention: one block per (window, head); 128 thr = 4 waves.
// ---------------------------------------------------------------------------
__global__ void k_win_attn(const unsigned short* __restrict__ qkvH,
                           unsigned short* __restrict__ attnH) {
  __shared__ __attribute__((aligned(16))) unsigned short Qa[64][72];
  __shared__ __attribute__((aligned(16))) unsigned short Ka[64][72];
  __shared__ __attribute__((aligned(16))) unsigned short Vt[64][72];
  __shared__ __attribute__((aligned(16))) unsigned short Pa[64][72];
  __shared__ float S[64][64];
  int blk = blockIdx.x;
  int h = blk % 12, w = blk / 12;
  int b = w >> 4, wi = w & 15, wy = wi >> 2, wx = wi & 3;
  int t = threadIdx.x;
#pragma unroll
  for (int r = 0; r < 4; ++r) {
    int cid = t + r * 128;
    int i = cid >> 3, co = (cid & 7) * 8;
    int gy = wy * 8 + (i >> 3), gx = wx * 8 + (i & 7);
    size_t row = (size_t)(b * 1024 + gy * 32 + gx) * 2304;
    *(uint4*)&Qa[i][co] = *(const uint4*)(qkvH + row + h * 64 + co);
    *(uint4*)&Ka[i][co] = *(const uint4*)(qkvH + row + 768 + h * 64 + co);
    uint4 dv = *(const uint4*)(qkvH + row + 1536 + h * 64 + co);
    const unsigned short* e = (const unsigned short*)&dv;
#pragma unroll
    for (int j = 0; j < 8; j++) Vt[co + j][i] = e[j];
  }
  __syncthreads();
  int wave = t >> 5, lane = t & 31, lh = lane >> 4, ll = lane & 15;
  int m0 = wave * 16;
  v8f zerov = {};
#pragma unroll
  for (int j = 0; j < 4; j++) {
    v8f a = zerov;
#pragma unroll
    for (int ks = 0; ks < 64; ks += 32) {
      FragBF bf, af;
      int n = j * 16 + ll;
      bf.q[0] = *(const uint4*)&Ka[n][ks + lh * 8];
      bf.q[1] = *(const uint4*)&Ka[n][ks + 16 + lh * 8];
      int m = m0 + ll;
      af.q[0] = *(const uint4*)&Qa[m][ks + lh * 8];
      af.q[1] = *(const uint4*)&Qa[m][ks + 16 + lh * 8];
      a = __builtin_amdgcn_wmma_f32_16x16x32_bf16(false, af.v, false, bf.v,
                                                  (short)0, a, false, false);
    }
#pragma unroll
    for (int r = 0; r < 8; r++) S[m0 + r + lh * 8][j * 16 + ll] = a[r] * 0.125f;
  }
  __syncthreads();
  if (t < 64) {
    float mx = -1e30f;
    for (int c = 0; c < 64; c++) mx = fmaxf(mx, S[t][c]);
    float sum = 0.f;
    for (int c = 0; c < 64; c++) { float e = __expf(S[t][c] - mx); S[t][c] = e; sum += e; }
    float inv = 1.f / sum;
    for (int c = 0; c < 64; c++) Pa[t][c] = f2bf(S[t][c] * inv);
  }
  __syncthreads();
#pragma unroll
  for (int j = 0; j < 4; j++) {
    v8f a = zerov;
#pragma unroll
    for (int ks = 0; ks < 64; ks += 32) {
      FragBF bf, af;
      int n = j * 16 + ll;
      bf.q[0] = *(const uint4*)&Vt[n][ks + lh * 8];
      bf.q[1] = *(const uint4*)&Vt[n][ks + 16 + lh * 8];
      int m = m0 + ll;
      af.q[0] = *(const uint4*)&Pa[m][ks + lh * 8];
      af.q[1] = *(const uint4*)&Pa[m][ks + 16 + lh * 8];
      a = __builtin_amdgcn_wmma_f32_16x16x32_bf16(false, af.v, false, bf.v,
                                                  (short)0, a, false, false);
    }
#pragma unroll
    for (int r = 0; r < 8; r++) {
      int iTok = m0 + r + lh * 8;
      int gy = wy * 8 + (iTok >> 3), gx = wx * 8 + (iTok & 7);
      size_t row = (size_t)(b * 1024 + gy * 32 + gx);
      attnH[row * 768 + h * 64 + j * 16 + ll] = f2bf(a[r]);
    }
  }
}

// ---------------------------------------------------------------------------
// Cross attention, flash-style online softmax over 16 KV tiles of 64.
// ---------------------------------------------------------------------------
__global__ void k_cross_attn(const unsigned short* __restrict__ qH,
                             const unsigned short* __restrict__ kvH,
                             unsigned short* __restrict__ outH) {
  __shared__ __attribute__((aligned(16))) unsigned short Qa[64][72];
  __shared__ __attribute__((aligned(16))) unsigned short Ka[64][72];
  __shared__ __attribute__((aligned(16))) unsigned short Vt[64][72];
  __shared__ __attribute__((aligned(16))) unsigned short Pa[64][72];
  __shared__ float S[64][64];
  __shared__ float mrow[64], lrow[64], arow[64];
  int blk = blockIdx.x;
  int b = blk / 192, rem = blk % 192;
  int h = rem >> 4, qt = rem & 15;
  int t = threadIdx.x;
#pragma unroll
  for (int r = 0; r < 4; ++r) {
    int cid = t + r * 128;
    int i = cid >> 3, co = (cid & 7) * 8;
    size_t row = (size_t)(b * 1024 + qt * 64 + i);
    *(uint4*)&Qa[i][co] = *(const uint4*)(qH + row * 768 + h * 64 + co);
  }
  if (t < 64) { mrow[t] = -1e30f; lrow[t] = 0.f; }
  __syncthreads();
  int wave = t >> 5, lane = t & 31, lh = lane >> 4, ll = lane & 15;
  int m0 = wave * 16;
  v8f zerov = {};
  v8f accO[4];
#pragma unroll
  for (int j = 0; j < 4; j++) accO[j] = zerov;

  for (int kt = 0; kt < 16; ++kt) {
#pragma unroll
    for (int r = 0; r < 4; ++r) {
      int cid = t + r * 128;
      int i = cid >> 3, co = (cid & 7) * 8;
      size_t row = (size_t)(b * 1024 + kt * 64 + i) * 1536;
      *(uint4*)&Ka[i][co] = *(const uint4*)(kvH + row + h * 64 + co);
      uint4 dv = *(const uint4*)(kvH + row + 768 + h * 64 + co);
      const unsigned short* e = (const unsigned short*)&dv;
#pragma unroll
      for (int j2 = 0; j2 < 8; j2++) Vt[co + j2][i] = e[j2];
    }
    __syncthreads();
#pragma unroll
    for (int j = 0; j < 4; j++) {
      v8f a = zerov;
#pragma unroll
      for (int ks = 0; ks < 64; ks += 32) {
        FragBF bf, af;
        int n = j * 16 + ll;
        bf.q[0] = *(const uint4*)&Ka[n][ks + lh * 8];
        bf.q[1] = *(const uint4*)&Ka[n][ks + 16 + lh * 8];
        int m = m0 + ll;
        af.q[0] = *(const uint4*)&Qa[m][ks + lh * 8];
        af.q[1] = *(const uint4*)&Qa[m][ks + 16 + lh * 8];
        a = __builtin_amdgcn_wmma_f32_16x16x32_bf16(false, af.v, false, bf.v,
                                                    (short)0, a, false, false);
      }
#pragma unroll
      for (int r = 0; r < 8; r++) S[m0 + r + lh * 8][j * 16 + ll] = a[r] * 0.125f;
    }
    __syncthreads();
    if (t < 64) {
      float mo = mrow[t], mt = mo;
      for (int c = 0; c < 64; c++) mt = fmaxf(mt, S[t][c]);
      float alpha = __expf(mo - mt);
      float s = 0.f;
      for (int c = 0; c < 64; c++) { float e = __expf(S[t][c] - mt); Pa[t][c] = f2bf(e); s += e; }
      lrow[t] = lrow[t] * alpha + s;
      mrow[t] = mt; arow[t] = alpha;
    }
    __syncthreads();
#pragma unroll
    for (int j = 0; j < 4; j++) {
#pragma unroll
      for (int r = 0; r < 8; r++) accO[j][r] *= arow[m0 + r + lh * 8];
#pragma unroll
      for (int ks = 0; ks < 64; ks += 32) {
        FragBF bf, af;
        int n = j * 16 + ll;
        bf.q[0] = *(const uint4*)&Vt[n][ks + lh * 8];
        bf.q[1] = *(const uint4*)&Vt[n][ks + 16 + lh * 8];
        int m = m0 + ll;
        af.q[0] = *(const uint4*)&Pa[m][ks + lh * 8];
        af.q[1] = *(const uint4*)&Pa[m][ks + 16 + lh * 8];
        accO[j] = __builtin_amdgcn_wmma_f32_16x16x32_bf16(false, af.v, false, bf.v,
                                                          (short)0, accO[j], false, false);
      }
    }
    __syncthreads();
  }
#pragma unroll
  for (int j = 0; j < 4; j++)
#pragma unroll
    for (int r = 0; r < 8; r++) {
      int m = m0 + r + lh * 8;
      size_t row = (size_t)(b * 1024 + qt * 64 + m);
      outH[row * 768 + h * 64 + j * 16 + ll] = f2bf(accO[j][r] / lrow[m]);
    }
}

// ---------------------------------------------------------------------------
// Spatial-attention gate pieces
// ---------------------------------------------------------------------------
__global__ void k_chanstat(const float* __restrict__ x, float* __restrict__ avg,
                           float* __restrict__ mx) {
  __shared__ float rs[256], rm[256];
  int row = blockIdx.x, t = threadIdx.x;
  const float* xr = x + (size_t)row * 768;
  float a = xr[t], b = xr[t + 256], c = xr[t + 512];
  rs[t] = a + b + c;
  rm[t] = fmaxf(a, fmaxf(b, c));
  __syncthreads();
  for (int o = 128; o > 0; o >>= 1) {
    if (t < o) { rs[t] += rs[t + o]; rm[t] = fmaxf(rm[t], rm[t + o]); }
    __syncthreads();
  }
  if (t == 0) { avg[row] = rs[0] * (1.f / 768.f); mx[row] = rm[0]; }
}

__global__ void k_sagate(const float* __restrict__ avg, const float* __restrict__ mx,
                         const float* __restrict__ sw, float* __restrict__ gate) {
  int i = blockIdx.x * 256 + threadIdx.x;
  if (i >= 8192) return;
  int b = i >> 10, g = i & 1023, y = g >> 5, x = g & 31;
  float acc = 0.f;
  for (int dy = 0; dy < 7; dy++) {
    int yy = y + dy - 3; if (yy < 0 || yy > 31) continue;
    for (int dx = 0; dx < 7; dx++) {
      int xx = x + dx - 3; if (xx < 0 || xx > 31) continue;
      int p = b * 1024 + yy * 32 + xx;
      acc += sw[dy * 7 + dx] * avg[p] + sw[49 + dy * 7 + dx] * mx[p];
    }
  }
  gate[i] = 1.f / (1.f + __expf(-acc));
}

__global__ void k_gate_apply(const float* __restrict__ x, const float* __restrict__ gate,
                             float* __restrict__ out) {
  size_t i = (size_t)blockIdx.x * 256 + threadIdx.x;
  if (i >= (size_t)8192 * 768) return;
  out[i] = x[i] * gate[i / 768];
}

// ---------------------------------------------------------------------------
// Host orchestration
// ---------------------------------------------------------------------------
template <int MODE>
static void launch_gemm(hipStream_t s, const unsigned short* A, const unsigned short* Bt,
                        const float* bias, const float* extra, float* outF,
                        unsigned short* outH, int M, int N, int K) {
  dim3 grid(N / 128, M / 128);
  k_gemm_bf16<MODE><<<grid, 256, 0, s>>>(A, Bt, bias, extra, outF, outH, M, N, K);
}

extern "C" void kernel_launch(void* const* d_in, const int* in_sizes, int n_in,
                              void* d_out, int out_size, void* d_ws, size_t ws_size,
                              hipStream_t stream) {
  (void)in_sizes; (void)n_in; (void)out_size; (void)ws_size;
  const float* rgb = (const float*)d_in[0];
  const float* depth = (const float*)d_in[1];
  const int RB = 2, DB = 17, XB = 32;
  auto P = [&](int i) { return (const float*)d_in[i]; };

  char* ws = (char*)d_ws; size_t off = 0;
  auto alloc = [&](size_t bytes) -> void* {
    void* p = ws + off; off += (bytes + 255) & ~(size_t)255; return p; };

  unsigned short* actH  = (unsigned short*)alloc(8192ull * 768 * 2);
  unsigned short* qkvH  = (unsigned short*)alloc(8192ull * 2304 * 2);
  unsigned short* attnH = (unsigned short*)alloc(8192ull * 768 * 2);
  unsigned short* hidH  = (unsigned short*)alloc(8192ull * 3072 * 2);
  unsigned short* tokH[2] = { (unsigned short*)alloc(8192ull * 768 * 2),
                              (unsigned short*)alloc(8192ull * 768 * 2) };
  float* xbuf[2] = { (float*)alloc(8192ull * 768 * 4), (float*)alloc(8192ull * 768 * 4) };
  unsigned short* qHb   = (unsigned short*)alloc(8192ull * 768 * 2);
  unsigned short* kvHb  = (unsigned short*)alloc(8192ull * 1536 * 2);
  unsigned short* cattH = (unsigned short*)alloc(8192ull * 768 * 2);
  float* cbao = (float*)alloc(8192ull * 768 * 4);
  float* avgb = (float*)alloc(8192 * 4);
  float* maxb = (float*)alloc(8192 * 4);
  float* gate = (float*)alloc(8192 * 4);
  unsigned short *peB[2], *qkvW[2], *projW[2], *fc1W[2], *fc2W[2];
  for (int s = 0; s < 2; s++) {
    peB[s]  = (unsigned short*)alloc(768ull * 768 * 2);
    qkvW[s] = (unsigned short*)alloc(4ull * 768 * 2304 * 2);
    projW[s]= (unsigned short*)alloc(4ull * 768 * 768 * 2);
    fc1W[s] = (unsigned short*)alloc(4ull * 768 * 3072 * 2);
    fc2W[s] = (unsigned short*)alloc(4ull * 3072 * 768 * 2);
  }
  unsigned short* qW  = (unsigned short*)alloc(768ull * 768 * 2);
  unsigned short* kvW = (unsigned short*)alloc(768ull * 1536 * 2);
  unsigned short* oW  = (unsigned short*)alloc(768ull * 768 * 2);

  auto conv = [&](const float* src, unsigned short* dst, size_t n) {
    k_f32_to_bf16<<<dim3((unsigned)((n + 255) / 256)), 256, 0, stream>>>(src, dst, (int)n);
  };
  auto convT = [&](const float* src, unsigned short* dst, int K, int N) {
    k_convT_bf16<<<dim3(N / 32, K / 32), 256, 0, stream>>>(src, dst, K, N);
  };

  for (int s = 0; s < 2; s++) {
    int base = s ? DB : RB;
    conv(P(base + 0), peB[s], 768ull * 768);   // pe_w already N-major
    for (int l = 0; l < 4; l++) {
      convT(P(base + 5)  + (size_t)l * 768 * 2304, qkvW[s] + (size_t)l * 768 * 2304, 768, 2304);
      convT(P(base + 7)  + (size_t)l * 768 * 768,  projW[s]+ (size_t)l * 768 * 768,  768, 768);
      convT(P(base + 11) + (size_t)l * 768 * 3072, fc1W[s] + (size_t)l * 768 * 3072, 768, 3072);
      convT(P(base + 13) + (size_t)l * 3072 * 768, fc2W[s] + (size_t)l * 3072 * 768, 3072, 768);
    }
  }
  convT(P(XB + 0), qW,  768, 768);
  convT(P(XB + 2), kvW, 768, 1536);
  convT(P(XB + 4), oW,  768, 768);

  for (int s = 0; s < 2; s++) {
    const float* img = s ? depth : rgb;
    int base = s ? DB : RB;
    float* x = xbuf[s];
    k_im2col<<<24576, 256, 0, stream>>>(img, actH);
    launch_gemm<3>(stream, actH, peB[s], P(base + 1), P(base + 2), x, nullptr, 8192, 768, 768);
    for (int l = 0; l < 4; l++) {
      k_layernorm_bf16<<<8192, 256, 0, stream>>>(x, P(base + 3) + l * 768, P(base + 4) + l * 768, actH);
      launch_gemm<0>(stream, actH, qkvW[s] + (size_t)l * 768 * 2304, P(base + 6) + l * 2304,
                     nullptr, nullptr, qkvH, 8192, 2304, 768);
      k_win_attn<<<8 * 16 * 12, 128, 0, stream>>>(qkvH, attnH);
      launch_gemm<2>(stream, attnH, projW[s] + (size_t)l * 768 * 768, P(base + 8) + l * 768,
                     x, x, nullptr, 8192, 768, 768);
      k_layernorm_bf16<<<8192, 256, 0, stream>>>(x, P(base + 9) + l * 768, P(base + 10) + l * 768, actH);
      launch_gemm<1>(stream, actH, fc1W[s] + (size_t)l * 768 * 3072, P(base + 12) + l * 3072,
                     nullptr, nullptr, hidH, 8192, 3072, 768);
      launch_gemm<2>(stream, hidH, fc2W[s] + (size_t)l * 3072 * 768, P(base + 14) + l * 768,
                     x, x, (l == 3) ? tokH[s] : nullptr, 8192, 768, 3072);
    }
  }
  launch_gemm<0>(stream, tokH[0], qW,  P(XB + 1), nullptr, nullptr, qHb,  8192, 768, 768);
  launch_gemm<0>(stream, tokH[1], kvW, P(XB + 3), nullptr, nullptr, kvHb, 8192, 1536, 768);
  k_cross_attn<<<8 * 12 * 16, 128, 0, stream>>>(qHb, kvHb, cattH);
  launch_gemm<0>(stream, cattH, oW, P(XB + 5), nullptr, cbao, nullptr, 8192, 768, 768);
  k_chanstat<<<8192, 256, 0, stream>>>(cbao, avgb, maxb);
  k_sagate<<<32, 256, 0, stream>>>(avgb, maxb, P(XB + 6), gate);
  k_gate_apply<<<24576, 256, 0, stream>>>(cbao, gate, (float*)d_out);
}
